// LapCluster_77661598646804
// MI455X (gfx1250) — compile-verified, matrix operand order checked by reading, not run on hardware
//
#include <hip/hip_runtime.h>

typedef __attribute__((ext_vector_type(16))) _Float16 v16h;
typedef __attribute__((ext_vector_type(8)))  _Float16 v8h;
typedef __attribute__((ext_vector_type(8)))  float    v8f;

#define DEV static __device__ __forceinline__

static constexpr int B = 8;
static constexpr int N = 16384;
static constexpr int WAVES = 8;              // 256-thread WG = 8 wave32
static constexpr int GRID_X = N / (WAVES * 16);  // 128 WGs per batch

// ---------------- fragment helpers (ISA 16-bit 16x32 A layout) ----------------
// lane r = row (M for A, N for B), hi = lane>=16 selects K-halves:
//   hi=0: K in {0..7} u {16..23}; hi=1: K in {8..15} u {24..31}
// -> two contiguous 8-half (16B) chunks per lane.

DEV v16h cat8(v8h lo, v8h hi) {
  return __builtin_shufflevector(lo, hi, 0,1,2,3,4,5,6,7,8,9,10,11,12,13,14,15);
}
DEV v8f vzero8f() { v8f z = {0.f,0.f,0.f,0.f,0.f,0.f,0.f,0.f}; return z; }

// A fragment: 16x32 tile of row-major [*, stride] matrix, tile base given.
DEV v16h frag_a(const _Float16* __restrict__ base, int stride, int r, int hi) {
  const _Float16* p = base + r * stride + 8 * hi;
  return cat8(*(const v8h*)p, *(const v8h*)(p + 16));
}
// B fragment: row = this lane's node feature row (N x K storage), k-tile base applied.
DEV v16h frag_b(const _Float16* __restrict__ row, int hi) {
  const _Float16* p = row + 8 * hi;
  return cat8(*(const v8h*)p, *(const v8h*)(p + 16));
}
DEV v8f wmma_f16(v16h a, v16h b, v8f c) {
  return __builtin_amdgcn_wmma_f32_16x16x32_f16(false, a, false, b, (short)0, c, false, false);
}
// bias + ReLU + cvt to f16; bias already offset to this lane's channel base.
DEV v8h bias_relu(v8f acc, const float* __restrict__ bias) {
  v8h r;
#pragma unroll
  for (int j = 0; j < 8; ++j) {
    float v = acc[j] + bias[j];
    r[j] = (_Float16)(v > 0.f ? v : 0.f);
  }
  return r;
}
// max for non-negative floats via unsigned-int atomic (IEEE order == uint order for >=0)
DEV void amax_nonneg(float* p, float v) {
  if (v > 0.f) atomicMax((unsigned int*)p, __float_as_uint(v));
}

// ---------------- small utility kernels ----------------
__global__ void k_zero(float* __restrict__ p, int n) {
  int i = blockIdx.x * blockDim.x + threadIdx.x;
  if (i < n) p[i] = 0.f;
}
__global__ void k_cvt(const float* __restrict__ s, _Float16* __restrict__ d, int n) {
  int i = blockIdx.x * blockDim.x + threadIdx.x;
  if (i < n) d[i] = (_Float16)s[i];
}
// w_in [256][28] -> f16 [256][32], K padded with zeros
__global__ void k_cvt_pad(const float* __restrict__ s, _Float16* __restrict__ d) {
  int i = blockIdx.x * blockDim.x + threadIdx.x;
  if (i < 256 * 32) {
    int m = i >> 5, k = i & 31;
    d[i] = (k < 28) ? (_Float16)s[m * 28 + k] : (_Float16)0.f;
  }
}

// ---------------- stage kernels ----------------
// F = relu(W_in x + b_in): x [B,N,28] f32 -> F [B,N,256] f16. M=256, K=32(padded).
__global__ __launch_bounds__(256) void k_in(const float* __restrict__ x,
                                            const _Float16* __restrict__ w,   // [256][32]
                                            const float* __restrict__ bias,
                                            _Float16* __restrict__ F) {
  int tid = threadIdx.x, lane = tid & 31, wave = tid >> 5;
  int r = lane & 15, hi = lane >> 4;
  int b = blockIdx.y;
  int node = (blockIdx.x * WAVES + wave) * 16 + r;
  const float* xr = x + ((size_t)b * N + node) * 28;
  v16h bf;
#pragma unroll
  for (int i = 0; i < 8; ++i) bf[i] = (_Float16)xr[8 * hi + i];
#pragma unroll
  for (int i = 0; i < 8; ++i) {
    int k = 16 + 8 * hi + i;
    bf[8 + i] = (k < 28) ? (_Float16)xr[k] : (_Float16)0.f;  // weight pad is zero anyway
  }
  _Float16* Frow = F + ((size_t)b * N + node) * 256;
#pragma unroll
  for (int mt = 0; mt < 16; ++mt) {
    v16h af = frag_a(w + mt * 16 * 32, 32, r, hi);
    v8f acc = vzero8f();
    acc = wmma_f16(af, bf, acc);
    *(v8h*)(Frow + mt * 16 + 8 * hi) = bias_relu(acc, bias + mt * 16 + 8 * hi);
  }
}

// Block front half: a = relu(w1 F) [64], g = relu(w2 a) [128] -> G; cluster max -> pool.
__global__ __launch_bounds__(256) void k_p(const _Float16* __restrict__ F,
                                           const _Float16* __restrict__ w1,   // [64][256]
                                           const float* __restrict__ b1,
                                           const _Float16* __restrict__ w2,   // [128][64]
                                           const float* __restrict__ b2,
                                           const int* __restrict__ idx,      // [B][N]
                                           _Float16* __restrict__ G,
                                           float* __restrict__ pool,         // [B][64][128]
                                           int nc) {
  __shared__ float spool[64 * 128];  // 32KB per-WG pool tile
  int tid = threadIdx.x, lane = tid & 31, wave = tid >> 5;
  int r = lane & 15, hi = lane >> 4;
  int b = blockIdx.y;
  int node = (blockIdx.x * WAVES + wave) * 16 + r;
  for (int t = tid; t < nc * 128; t += 256) spool[t] = 0.f;
  __syncthreads();

  const _Float16* Frow = F + ((size_t)b * N + node) * 256;
  v8f acc1[4];
#pragma unroll
  for (int mt = 0; mt < 4; ++mt) acc1[mt] = vzero8f();
#pragma unroll
  for (int kt = 0; kt < 8; ++kt) {                 // K=256
    v16h bf = frag_b(Frow + kt * 32, hi);
#pragma unroll
    for (int mt = 0; mt < 4; ++mt)                 // M=64
      acc1[mt] = wmma_f16(frag_a(w1 + (mt * 16) * 256 + kt * 32, 256, r, hi), bf, acc1[mt]);
  }
  // D-fragments become next B-fragments in-register (lane=node, 8 contiguous channels)
  v16h bf2[2];
#pragma unroll
  for (int k2 = 0; k2 < 2; ++k2) {
    v8h lo = bias_relu(acc1[2 * k2],     b1 + 32 * k2 + 8 * hi);
    v8h hb = bias_relu(acc1[2 * k2 + 1], b1 + 32 * k2 + 16 + 8 * hi);
    bf2[k2] = cat8(lo, hb);
  }
  int c = idx[(size_t)b * N + node];
  _Float16* Grow = G + ((size_t)b * N + node) * 128;
#pragma unroll
  for (int mt = 0; mt < 8; ++mt) {                 // M=128, K=64
    v8f acc = vzero8f();
#pragma unroll
    for (int k2 = 0; k2 < 2; ++k2)
      acc = wmma_f16(frag_a(w2 + (mt * 16) * 64 + k2 * 32, 64, r, hi), bf2[k2], acc);
    v8h o = bias_relu(acc, b2 + mt * 16 + 8 * hi);
    *(v8h*)(Grow + mt * 16 + 8 * hi) = o;
    float* sp = spool + c * 128 + mt * 16 + 8 * hi;
#pragma unroll
    for (int j = 0; j < 8; ++j) amax_nonneg(sp + j, (float)o[j]);
  }
  __syncthreads();
  float* gp = pool + (size_t)b * 64 * 128;
  for (int t = tid; t < nc * 128; t += 256) amax_nonneg(gp + t, spool[t]);
}

// Block back half: f3 = relu(w3 g) [128]; F = concat(f3, pool[idx]) [256].
__global__ __launch_bounds__(256) void k_q(const _Float16* __restrict__ G,
                                           const _Float16* __restrict__ w3,   // [128][128]
                                           const float* __restrict__ b3,
                                           const int* __restrict__ idx,
                                           const float* __restrict__ pool,   // [B][64][128]
                                           _Float16* __restrict__ F) {
  int tid = threadIdx.x, lane = tid & 31, wave = tid >> 5;
  int r = lane & 15, hi = lane >> 4;
  int b = blockIdx.y;
  int node = (blockIdx.x * WAVES + wave) * 16 + r;
  const _Float16* Grow = G + ((size_t)b * N + node) * 128;
  v8f acc[8];
#pragma unroll
  for (int mt = 0; mt < 8; ++mt) acc[mt] = vzero8f();
#pragma unroll
  for (int kt = 0; kt < 4; ++kt) {                 // K=128
    v16h bf = frag_b(Grow + kt * 32, hi);
#pragma unroll
    for (int mt = 0; mt < 8; ++mt)                 // M=128
      acc[mt] = wmma_f16(frag_a(w3 + (mt * 16) * 128 + kt * 32, 128, r, hi), bf, acc[mt]);
  }
  _Float16* Frow = F + ((size_t)b * N + node) * 256;
  int c = idx[(size_t)b * N + node];
  const float* prow = pool + ((size_t)b * 64 + c) * 128;
#pragma unroll
  for (int mt = 0; mt < 8; ++mt) {
    *(v8h*)(Frow + mt * 16 + 8 * hi) = bias_relu(acc[mt], b3 + mt * 16 + 8 * hi);
    v8h ph;
#pragma unroll
    for (int j = 0; j < 8; ++j) ph[j] = (_Float16)prow[mt * 16 + 8 * hi + j];
    *(v8h*)(Frow + 128 + mt * 16 + 8 * hi) = ph;   // concat second half
  }
}

// o1 = relu(wo1 F), o2 = relu(wo2 o1), out[b][c] = max over N.
__global__ __launch_bounds__(256) void k_out(const _Float16* __restrict__ F,
                                             const _Float16* __restrict__ wo1,  // [128][256]
                                             const float* __restrict__ bo1,
                                             const _Float16* __restrict__ wo2,  // [128][128]
                                             const float* __restrict__ bo2,
                                             float* __restrict__ out) {
  int tid = threadIdx.x, lane = tid & 31, wave = tid >> 5;
  int r = lane & 15, hi = lane >> 4;
  int b = blockIdx.y;
  int node = (blockIdx.x * WAVES + wave) * 16 + r;
  const _Float16* Frow = F + ((size_t)b * N + node) * 256;
  v8f acc1[8];
#pragma unroll
  for (int mt = 0; mt < 8; ++mt) acc1[mt] = vzero8f();
#pragma unroll
  for (int kt = 0; kt < 8; ++kt) {                 // K=256
    v16h bf = frag_b(Frow + kt * 32, hi);
#pragma unroll
    for (int mt = 0; mt < 8; ++mt)                 // M=128
      acc1[mt] = wmma_f16(frag_a(wo1 + (mt * 16) * 256 + kt * 32, 256, r, hi), bf, acc1[mt]);
  }
  v8f acc2[8];
#pragma unroll
  for (int mt = 0; mt < 8; ++mt) acc2[mt] = vzero8f();
#pragma unroll
  for (int k2 = 0; k2 < 4; ++k2) {                 // K=128
    v8h lo = bias_relu(acc1[2 * k2],     bo1 + 32 * k2 + 8 * hi);
    v8h hb = bias_relu(acc1[2 * k2 + 1], bo1 + 32 * k2 + 16 + 8 * hi);
    v16h bf2 = cat8(lo, hb);
#pragma unroll
    for (int mt = 0; mt < 8; ++mt)
      acc2[mt] = wmma_f16(frag_a(wo2 + (mt * 16) * 128 + k2 * 32, 128, r, hi), bf2, acc2[mt]);
  }
  float* ob = out + b * 128;
#pragma unroll
  for (int mt = 0; mt < 8; ++mt) {
#pragma unroll
    for (int j = 0; j < 8; ++j) {
      float v = acc2[mt][j] + bo2[mt * 16 + 8 * hi + j];
      v = v > 0.f ? v : 0.f;
      // butterfly max over the 16 nodes of this tile (stays within half-wave: xor<16)
      for (int m = 1; m < 16; m <<= 1) v = fmaxf(v, __shfl_xor(v, m, 32));
      if (r == 0) amax_nonneg(ob + mt * 16 + 8 * hi + j, v);
    }
  }
}

// ---------------- host side ----------------
extern "C" void kernel_launch(void* const* d_in, const int* in_sizes, int n_in,
                              void* d_out, int out_size, void* d_ws, size_t ws_size,
                              hipStream_t stream) {
  const float* x        = (const float*)d_in[0];
  const int*   clusters = (const int*)d_in[1];
  const float* w_in     = (const float*)d_in[2];
  const float* b_in     = (const float*)d_in[3];
  const float* w1       = (const float*)d_in[4];
  const float* b1       = (const float*)d_in[5];
  const float* w2       = (const float*)d_in[6];
  const float* b2       = (const float*)d_in[7];
  const float* w3       = (const float*)d_in[8];
  const float* b3       = (const float*)d_in[9];
  const float* wo1      = (const float*)d_in[10];
  const float* bo1      = (const float*)d_in[11];
  const float* wo2      = (const float*)d_in[12];
  const float* bo2      = (const float*)d_in[13];
  float* out = (float*)d_out;

  // workspace layout (bytes)
  char* ws = (char*)d_ws;
  size_t off = 0;
  _Float16* F = (_Float16*)(ws + off); off += (size_t)B * N * 256 * 2;   // 64 MB
  _Float16* G = (_Float16*)(ws + off); off += (size_t)B * N * 128 * 2;   // 32 MB
  float* poolAll = (float*)(ws + off); off += (size_t)3 * B * 64 * 128 * 4;
  _Float16* w_in16 = (_Float16*)(ws + off); off += (size_t)256 * 32 * 2;
  _Float16* w1_16  = (_Float16*)(ws + off); off += (size_t)3 * 64 * 256 * 2;
  _Float16* w2_16  = (_Float16*)(ws + off); off += (size_t)3 * 128 * 64 * 2;
  _Float16* w3_16  = (_Float16*)(ws + off); off += (size_t)3 * 128 * 128 * 2;
  _Float16* wo1_16 = (_Float16*)(ws + off); off += (size_t)128 * 256 * 2;
  _Float16* wo2_16 = (_Float16*)(ws + off); off += (size_t)128 * 128 * 2;

  // weight conversion (cheap, every launch: deterministic, capture-safe)
  k_cvt_pad<<<(256 * 32 + 255) / 256, 256, 0, stream>>>(w_in, w_in16);
  k_cvt<<<(3 * 64 * 256 + 255) / 256, 256, 0, stream>>>(w1, w1_16, 3 * 64 * 256);
  k_cvt<<<(3 * 128 * 64 + 255) / 256, 256, 0, stream>>>(w2, w2_16, 3 * 128 * 64);
  k_cvt<<<(3 * 128 * 128 + 255) / 256, 256, 0, stream>>>(w3, w3_16, 3 * 128 * 128);
  k_cvt<<<(128 * 256 + 255) / 256, 256, 0, stream>>>(wo1, wo1_16, 128 * 256);
  k_cvt<<<(128 * 128 + 255) / 256, 256, 0, stream>>>(wo2, wo2_16, 128 * 128);

  // zero pool accumulators and output (atomic-max targets, nonneg data)
  k_zero<<<(3 * B * 64 * 128 + 255) / 256, 256, 0, stream>>>(poolAll, 3 * B * 64 * 128);
  k_zero<<<(B * 128 + 255) / 256, 256, 0, stream>>>(out, B * 128);

  dim3 grid(GRID_X, B);
  k_in<<<grid, 256, 0, stream>>>(x, w_in16, b_in, F);

  const int NCs[3] = {16, 32, 64};
  for (int i = 0; i < 3; ++i) {
    const int* idx = clusters + (size_t)i * B * N;
    float* pool = poolAll + (size_t)i * B * 64 * 128;
    k_p<<<grid, 256, 0, stream>>>(F, w1_16 + (size_t)i * 64 * 256, b1 + i * 64,
                                  w2_16 + (size_t)i * 128 * 64, b2 + i * 128,
                                  idx, G, pool, NCs[i]);
    k_q<<<grid, 256, 0, stream>>>(G, w3_16 + (size_t)i * 128 * 128, b3 + i * 128,
                                  idx, pool, F);
  }
  k_out<<<grid, 256, 0, stream>>>(F, wo1_16, bo1, wo2_16, bo2, out);
}
